// BiLSTM_CRF_6708738916988
// MI455X (gfx1250) — matrix operand order repaired and checked by reading, not compiled, not run on hardware
//
#include <hip/hip_runtime.h>
#include <hip/hip_bf16.h>

// ---------------------------------------------------------------------------
// BiLSTM-CRF for MI455X (gfx1250, wave32, WMMA)
//   Phase 0: convert weights / gather embeddings to bf16
//   Phase 1: WMMA bf16 GEMM  Gx[dir][t][r] = X @ W_ih^T + b   (34 GFLOP)
//            2x2 register-blocked: 4 WMMAs per 8 b128 loads per k-step
//   Phase 2: cooperative recurrence: 8 WGPs/direction, W_hh in LDS (bf16),
//            one device-scope barrier per step, double-buffered gates,
//            global_prefetch of next step's Gx row
//   Phase 3: output projection to 5 tags
//   Phase 4: single-wave Viterbi, backpointers in LDS
// ---------------------------------------------------------------------------

constexpr int T_SEQ = 4096;
constexpr int E_DIM = 1024;
constexpr int H_DIM = 512;
constexpr int G4    = 2048;   // 4*H
constexpr int TAGS  = 5;
constexpr int NWG_DIR = 8;    // workgroups cooperating per direction

typedef __bf16 bf16x8  __attribute__((ext_vector_type(8)));
typedef __bf16 bf16x16 __attribute__((ext_vector_type(16)));
typedef float  f32x8   __attribute__((ext_vector_type(8)));

// ---- f32 -> bf16 with round-to-nearest-even -------------------------------
__device__ __forceinline__ unsigned short f2bf(float f) {
    unsigned u = __float_as_uint(f);
    unsigned rnd = 0x7FFFu + ((u >> 16) & 1u);
    return (unsigned short)((u + rnd) >> 16);
}

__global__ void cvt_f32_bf16(const float* __restrict__ src,
                             unsigned short* __restrict__ dst, int n) {
    int i = blockIdx.x * blockDim.x + threadIdx.x;
    if (i < n) dst[i] = f2bf(src[i]);
}

// ---- embedding gather (one block per token row) ---------------------------
__global__ void gather_embed(const int* __restrict__ sent,
                             const float* __restrict__ embed,
                             unsigned short* __restrict__ X) {
    int t = blockIdx.x;
    long row = (long)sent[t];
    const float* src = embed + row * E_DIM;
    unsigned short* dst = X + (long)t * E_DIM;
    for (int e = threadIdx.x; e < E_DIM; e += blockDim.x) dst[e] = f2bf(src[e]);
}

// ---- WMMA fragment loaders ------------------------------------------------
// A (16x32 bf16): lanes 0-15 carry K = 0..7 & 16..23; lanes 16-31 K = 8..15 & 24..31
__device__ __forceinline__ bf16x16 ld_afrag(const unsigned short* row,
                                            int kb, int hi) {
    bf16x8 a0 = *(const bf16x8*)(row + kb + hi * 8);
    bf16x8 a1 = *(const bf16x8*)(row + kb + 16 + hi * 8);
    return __builtin_shufflevector(a0, a1,
        0,1,2,3,4,5,6,7,8,9,10,11,12,13,14,15);
}
// B (32x16 bf16, K x N): lanes 0-15 K = 0..7 & 16..23; lanes 16-31 K = 8..15 & 24..31
__device__ __forceinline__ bf16x16 ld_bfrag(const unsigned short* row,
                                            int kb, int hi) {
    bf16x8 b0 = *(const bf16x8*)(row + kb + hi * 8);
    bf16x8 b1 = *(const bf16x8*)(row + kb + 16 + hi * 8);
    return __builtin_shufflevector(b0, b1,
        0,1,2,3,4,5,6,7,8,9,10,11,12,13,14,15);
}

// ---- Phase 1: WMMA GEMM, 2x2 register blocking ----------------------------
// Each wave computes a 32x32 block of C[t][ncol]; ncol in [0,4096): dir = ncol>>11.
// (dir boundary 2048 is a multiple of 32, so a block never straddles directions.)
__global__ void wmma_input_gemm(const unsigned short* __restrict__ X,
                                const unsigned short* __restrict__ WihF,
                                const unsigned short* __restrict__ WihB,
                                const float* __restrict__ bF,
                                const float* __restrict__ bB,
                                float* __restrict__ Gx) {
    const int lane = threadIdx.x & 31;
    const int wave = threadIdx.x >> 5;
    const int superId = blockIdx.x * 8 + wave;      // 16384 32x32 blocks
    const int superN = (2 * G4) / 32;               // 128
    const int sm = superId / superN;
    const int sn = superId % superN;

    const int l15 = lane & 15;
    const int hi  = lane >> 4;

    const unsigned short* arow0 = X + (long)(sm * 32 + l15) * E_DIM;
    const unsigned short* arow1 = arow0 + 16 * E_DIM;

    const int dir = (sn * 32) >> 11;
    const int r0  = (sn * 32 + l15) & (G4 - 1);     // column of B0 fragment
    const unsigned short* wbase = dir ? WihB : WihF;
    const unsigned short* wrow0 = wbase + (long)r0 * E_DIM;
    const unsigned short* wrow1 = wrow0 + 16 * E_DIM;   // column r0+16

    f32x8 c00 = {}, c01 = {}, c10 = {}, c11 = {};
    for (int kb = 0; kb < E_DIM; kb += 32) {
        bf16x16 A0 = ld_afrag(arow0, kb, hi);
        bf16x16 A1 = ld_afrag(arow1, kb, hi);
        bf16x16 B0 = ld_bfrag(wrow0, kb, hi);
        bf16x16 B1 = ld_bfrag(wrow1, kb, hi);
        c00 = __builtin_amdgcn_wmma_f32_16x16x32_bf16(
                  false, A0, false, B0, (short)0, c00, false, false);
        c01 = __builtin_amdgcn_wmma_f32_16x16x32_bf16(
                  false, A0, false, B1, (short)0, c01, false, false);
        c10 = __builtin_amdgcn_wmma_f32_16x16x32_bf16(
                  false, A1, false, B0, (short)0, c10, false, false);
        c11 = __builtin_amdgcn_wmma_f32_16x16x32_bf16(
                  false, A1, false, B1, (short)0, c11, false, false);
    }

    // C layout per tile: lane n = l15, rows m = hi*8 + j in element j
    const float* bias = dir ? bB : bF;
    const float bias0 = bias[r0];
    const float bias1 = bias[r0 + 16];
    const int mbase = hi * 8;
    float* GxD = Gx + (long)dir * T_SEQ * G4;
#pragma unroll
    for (int j = 0; j < 8; ++j) {
        const int tA = sm * 32 + mbase + j;        // rows of A0 tiles
        const int tB = tA + 16;                    // rows of A1 tiles
        GxD[(long)tA * G4 + r0]      = c00[j] + bias0;
        GxD[(long)tA * G4 + r0 + 16] = c01[j] + bias1;
        GxD[(long)tB * G4 + r0]      = c10[j] + bias0;
        GxD[(long)tB * G4 + r0 + 16] = c11[j] + bias1;
    }
}

// ---- barrier counter init -------------------------------------------------
__global__ void init_counters(unsigned* cnt) {
    if (threadIdx.x < 2) cnt[threadIdx.x] = 0u;
}

// ---- Phase 2: cooperative LSTM recurrence ---------------------------------
// grid = 16 blocks x 256 threads. blocks [0,8) = forward dir, [8,16) = backward.
// Each block owns 256 rows of W_hh resident in LDS (bf16, stride-257 dwords
// to avoid bank conflicts). One device-scope barrier per step; gates are
// double-buffered in L2; every block redundantly does the cell update so no
// second barrier is needed.
__global__ void lstm_recurrence(const float* __restrict__ Gx,
                                const unsigned short* __restrict__ WhhF,
                                const unsigned short* __restrict__ WhhB,
                                const float* __restrict__ h0,
                                const float* __restrict__ c0,
                                float* __restrict__ hs,
                                float* __restrict__ gates,
                                unsigned* __restrict__ cnt) {
    const int dir = blockIdx.x / NWG_DIR;
    const int wg  = blockIdx.x % NWG_DIR;
    const int tid = threadIdx.x;                 // 256 threads

    extern __shared__ char smem[];
    unsigned* w_lds = (unsigned*)smem;                       // 256 * 257 dwords
    float* hsh = (float*)(smem + 256 * 257 * 4);             // h[512]
    float* csh = hsh + H_DIM;                                // c[512]

    const int r = wg * 256 + tid;                // global gate row this thread owns

    // load my W_hh row into LDS (512 bf16 = 256 dwords), padded stride 257
    const unsigned* wsrc =
        (const unsigned*)(dir ? WhhB : WhhF) + (long)r * (H_DIM / 2);
    unsigned* wdst = w_lds + tid * 257;
    for (int k = 0; k < H_DIM / 2; ++k) wdst[k] = wsrc[k];

    for (int j = tid; j < H_DIM; j += 256) {
        hsh[j] = h0[dir * H_DIM + j];
        csh[j] = c0[dir * H_DIM + j];
    }
    __syncthreads();

    unsigned* mycnt = cnt + dir;
    const float* GxD = Gx + (long)dir * T_SEQ * G4;
    float* hsD = hs + (long)dir * T_SEQ * H_DIM;
    float* gbufs = gates + (long)dir * 2 * G4;

    for (int step = 0; step < T_SEQ; ++step) {
        const int t = dir ? (T_SEQ - 1 - step) : step;

        // prefetch next step's Gx element (emits global_prefetch_b8)
        if (step + 1 < T_SEQ) {
            const int tn = dir ? (t - 1) : (t + 1);
            __builtin_prefetch(&GxD[(long)tn * G4 + r], 0, 1);
        }

        // gate[r] = Gx[t][r] + W_hh[r] . h
        float acc = GxD[(long)t * G4 + r];
        const unsigned* wr = w_lds + tid * 257;
#pragma unroll 4
        for (int k2 = 0; k2 < H_DIM / 2; ++k2) {
            unsigned w = wr[k2];
            float w0 = __uint_as_float(w << 16);
            float w1 = __uint_as_float(w & 0xFFFF0000u);
            acc = fmaf(w0, hsh[2 * k2], acc);
            acc = fmaf(w1, hsh[2 * k2 + 1], acc);
        }
        float* gb = gbufs + (step & 1) * G4;
        gb[r] = acc;

        // ---- device-scope barrier across the 8 blocks of this direction ----
        __threadfence();
        __syncthreads();
        if (tid == 0) {
            atomicAdd(mycnt, 1u);
            const unsigned target = (unsigned)(step + 1) * NWG_DIR;
            while (__hip_atomic_load(mycnt, __ATOMIC_RELAXED,
                                     __HIP_MEMORY_SCOPE_AGENT) < target) {
                __builtin_amdgcn_s_sleep(1);
            }
        }
        __syncthreads();
        __threadfence();

        // ---- redundant cell update (all blocks compute full h,c) ----
#pragma unroll
        for (int e = 0; e < 2; ++e) {
            const int j = tid + e * 256;
            float ig = gb[j];
            float fg = gb[H_DIM + j];
            float gg = gb[2 * H_DIM + j];
            float og = gb[3 * H_DIM + j];
            float is = 1.0f / (1.0f + __expf(-ig));
            float fs = 1.0f / (1.0f + __expf(-fg));
            float gt = tanhf(gg);
            float os = 1.0f / (1.0f + __expf(-og));
            float cn = fs * csh[j] + is * gt;
            float hn = os * tanhf(cn);
            csh[j] = cn;
            hsh[j] = hn;
            if (wg == 0) hsD[(long)t * H_DIM + j] = hn;
        }
        __syncthreads();
    }
}

// ---- Phase 3: output projection to 5 tags ---------------------------------
__global__ void output_proj(const float* __restrict__ hs,
                            const float* __restrict__ Wout,
                            const float* __restrict__ bout,
                            float* __restrict__ feats) {
    int idx = blockIdx.x * blockDim.x + threadIdx.x;
    if (idx >= T_SEQ * TAGS) return;
    int t = idx / TAGS, tag = idx % TAGS;
    const float* hf = hs + (long)t * H_DIM;
    const float* hb = hs + (long)T_SEQ * H_DIM + (long)t * H_DIM;
    const float* w  = Wout + tag * (2 * H_DIM);
    float s = bout[tag];
#pragma unroll 4
    for (int k = 0; k < H_DIM; ++k) s = fmaf(hf[k], w[k], s);
#pragma unroll 4
    for (int k = 0; k < H_DIM; ++k) s = fmaf(hb[k], w[H_DIM + k], s);
    feats[idx] = s;
}

// ---- Phase 4: Viterbi (single wave, backpointers in LDS) ------------------
__global__ void viterbi(const float* __restrict__ feats,
                        const float* __restrict__ trans,
                        float* __restrict__ out) {
    __shared__ float fv[8];
    __shared__ float tr[TAGS * TAGS];
    __shared__ float fchunk[32 * TAGS];
    __shared__ unsigned char bp[T_SEQ * TAGS];   // 20 KB backpointers in LDS

    const int tid = threadIdx.x;                 // 32 threads = 1 wave
    if (tid < TAGS * TAGS) tr[tid] = trans[tid];
    if (tid < TAGS) fv[tid] = (tid == 3) ? 0.0f : -10000.0f;   // START = 3
    __syncthreads();

    for (int blk = 0; blk < T_SEQ / 32; ++blk) {
        // stage 32 steps (160 floats) of features
#pragma unroll
        for (int j = 0; j < TAGS; ++j) {
            int i = j * 32 + tid;
            fchunk[i] = feats[blk * 32 * TAGS + i];
        }
        __syncthreads();
        for (int s = 0; s < 32; ++s) {
            float nf = 0.0f; int bi = 0;
            if (tid < TAGS) {
                float best = -3.4e38f;
#pragma unroll
                for (int p = 0; p < TAGS; ++p) {
                    float sc = fv[p] + tr[tid * TAGS + p];
                    if (sc > best) { best = sc; bi = p; }      // first-max wins
                }
                nf = best + fchunk[s * TAGS + tid];
            }
            __syncthreads();
            if (tid < TAGS) {
                fv[tid] = nf;
                bp[(blk * 32 + s) * TAGS + tid] = (unsigned char)bi;
            }
            __syncthreads();
        }
    }

    if (tid == 0) {
        // terminal = fv + transitions[STOP][:]   (STOP = 4)
        float best = -3.4e38f; int bi = 0;
        for (int p = 0; p < TAGS; ++p) {
            float sc = fv[p] + tr[4 * TAGS + p];
            if (sc > best) { best = sc; bi = p; }
        }
        out[0] = best;                       // score
        int y = bi;
        out[1 + T_SEQ - 1] = (float)y;       // path[T-1]
        for (int t = T_SEQ - 1; t >= 1; --t) {
            y = bp[t * TAGS + y];
            out[t] = (float)y;               // path[t-1] at out[1+(t-1)]
        }
    }
}

// ---------------------------------------------------------------------------
extern "C" void kernel_launch(void* const* d_in, const int* in_sizes, int n_in,
                              void* d_out, int out_size, void* d_ws, size_t ws_size,
                              hipStream_t stream) {
    (void)in_sizes; (void)n_in; (void)out_size; (void)ws_size;

    const int*   sentence = (const int*)  d_in[0];
    const float* embed    = (const float*)d_in[1];
    const float* W_ih_f   = (const float*)d_in[2];
    const float* W_hh_f   = (const float*)d_in[3];
    const float* b_f      = (const float*)d_in[4];
    const float* W_ih_b   = (const float*)d_in[5];
    const float* W_hh_b   = (const float*)d_in[6];
    const float* b_b      = (const float*)d_in[7];
    const float* h0       = (const float*)d_in[8];
    const float* c0       = (const float*)d_in[9];
    const float* W_out    = (const float*)d_in[10];
    const float* b_out    = (const float*)d_in[11];
    const float* transitions = (const float*)d_in[12];
    float* out = (float*)d_out;

    // ---- workspace carve-up (256B aligned) ----
    char* ws = (char*)d_ws;
    size_t off = 0;
    auto carve = [&](size_t bytes) -> char* {
        char* p = ws + off;
        off = (off + bytes + 255) & ~(size_t)255;
        return p;
    };
    unsigned short* Xbf   = (unsigned short*)carve((size_t)T_SEQ * E_DIM * 2);
    unsigned short* WihFb = (unsigned short*)carve((size_t)G4 * E_DIM * 2);
    unsigned short* WihBb = (unsigned short*)carve((size_t)G4 * E_DIM * 2);
    unsigned short* WhhFb = (unsigned short*)carve((size_t)G4 * H_DIM * 2);
    unsigned short* WhhBb = (unsigned short*)carve((size_t)G4 * H_DIM * 2);
    float* Gx    = (float*)carve((size_t)2 * T_SEQ * G4 * 4);
    float* hs    = (float*)carve((size_t)2 * T_SEQ * H_DIM * 4);
    float* gates = (float*)carve((size_t)2 * 2 * G4 * 4);
    float* feats = (float*)carve((size_t)T_SEQ * TAGS * 4);
    unsigned* cnt = (unsigned*)carve(64);

    // Phase 0: barrier init + bf16 conversions + embedding gather
    init_counters<<<1, 32, 0, stream>>>(cnt);
    {
        int n1 = G4 * E_DIM;   // W_ih
        int n2 = G4 * H_DIM;   // W_hh
        cvt_f32_bf16<<<(n1 + 255) / 256, 256, 0, stream>>>(W_ih_f, WihFb, n1);
        cvt_f32_bf16<<<(n1 + 255) / 256, 256, 0, stream>>>(W_ih_b, WihBb, n1);
        cvt_f32_bf16<<<(n2 + 255) / 256, 256, 0, stream>>>(W_hh_f, WhhFb, n2);
        cvt_f32_bf16<<<(n2 + 255) / 256, 256, 0, stream>>>(W_hh_b, WhhBb, n2);
    }
    gather_embed<<<T_SEQ, 256, 0, stream>>>(sentence, embed, Xbf);

    // Phase 1: WMMA GEMM (16384 32x32 blocks, 8 waves/block)
    {
        const int blocks32 = (T_SEQ / 32) * ((2 * G4) / 32);   // 16384
        wmma_input_gemm<<<blocks32 / 8, 256, 0, stream>>>(Xbf, WihFb, WihBb,
                                                          b_f, b_b, Gx);
    }

    // Phase 2: cooperative recurrence (16 blocks, ~261 KB dynamic LDS each)
    {
        const int ldsBytes = 256 * 257 * 4 + 2 * H_DIM * 4;  // 267264
        hipFuncSetAttribute((const void*)lstm_recurrence,
                            hipFuncAttributeMaxDynamicSharedMemorySize, ldsBytes);
        lstm_recurrence<<<2 * NWG_DIR, 256, ldsBytes, stream>>>(
            Gx, WhhFb, WhhBb, h0, c0, hs, gates, cnt);
    }

    // Phase 3: projection to tag space
    {
        const int n = T_SEQ * TAGS;
        output_proj<<<(n + 255) / 256, 256, 0, stream>>>(hs, W_out, b_out, feats);
    }

    // Phase 4: Viterbi decode (single wave)
    viterbi<<<1, 32, 0, stream>>>(feats, transitions, out);
}
